// MultiHeadAttentionBlock_60421599920455
// MI455X (gfx1250) — matrix operand order repaired and checked by reading, not compiled, not run on hardware
//
#include <hip/hip_runtime.h>
#include <hip/hip_bf16.h>
#include <math.h>
#include <stdint.h>

// ---------------- types ----------------
typedef __attribute__((ext_vector_type(16))) __bf16 bf16x16;
typedef __attribute__((ext_vector_type(8)))  __bf16 bf16x8;
typedef __attribute__((ext_vector_type(2)))  __bf16 bf16x2;
typedef __attribute__((ext_vector_type(8)))  float  f32x8;
typedef __attribute__((ext_vector_type(4)))  int    i32x4;

#define B_      8
#define D_      4096
#define H_      32
#define KV_     8
#define HD_     128
#define MS_     4096
#define CH_     64
#define NCHUNK_ (MS_ / CH_)     // 64 chunks -> 4096 workgroups for the cache stream

// ---- gfx1250 async global->LDS (probe-verified signature: int4 AS1* / AS3*) ----
#if __has_builtin(__builtin_amdgcn_global_load_async_to_lds_b128)
#define ASYNC_LDS 1
typedef __attribute__((address_space(1))) i32x4 gi4_t;
typedef __attribute__((address_space(3))) i32x4 li4_t;
#else
#define ASYNC_LDS 0
#endif
#if __has_builtin(__builtin_amdgcn_s_wait_asynccnt)
#define WAIT_ASYNC() __builtin_amdgcn_s_wait_asynccnt(0)
#else
#define WAIT_ASYNC()
#endif

__device__ __forceinline__ f32x8 wmma_bf16(bf16x16 a, bf16x16 b, f32x8 c) {
  return __builtin_amdgcn_wmma_f32_16x16x32_bf16(false, a, false, b, (short)0, c,
                                                 false, false);
}

__device__ __forceinline__ bf16x16 pack16(bf16x8 lo, bf16x8 hi) {
  bf16x16 r;
#pragma unroll
  for (int i = 0; i < 8; ++i) { r[i] = lo[i]; r[i + 8] = hi[i]; }
  return r;
}

__device__ __forceinline__ f32x8 zero8f() {
  f32x8 r;
#pragma unroll
  for (int i = 0; i < 8; ++i) r[i] = 0.f;
  return r;
}

// 16 floats -> bf16x16 via packed converts (v_cvt_pk_bf16_f32)
__device__ __forceinline__ bf16x16 cvt16_pk(const float (&t)[16]) {
  union { bf16x16 v; bf16x2 h[8]; } u;
#if __has_builtin(__builtin_amdgcn_cvt_pk_bf16_f32)
#pragma unroll
  for (int i = 0; i < 8; ++i) {
    auto pk = __builtin_amdgcn_cvt_pk_bf16_f32(t[2 * i], t[2 * i + 1]);
    __builtin_memcpy(&u.h[i], &pk, 4);
  }
#else
#pragma unroll
  for (int i = 0; i < 16; ++i) u.v[i] = (__bf16)t[i];
#endif
  return u.v;
}

// convert 16 contiguous fp32 (16B-aligned) into a bf16x16 fragment run
__device__ __forceinline__ bf16x16 cvt16_f32(const float* p) {
  const float4* q = (const float4*)p;
  float4 a = q[0], b = q[1], c = q[2], d = q[3];
  float t[16] = {a.x, a.y, a.z, a.w, b.x, b.y, b.z, b.w,
                 c.x, c.y, c.z, c.w, d.x, d.y, d.z, d.w};
  return cvt16_pk(t);
}

__device__ __forceinline__ void fetch_w(const float* __restrict__ W, int row,
                                        int ncol, int N, float (&wa)[16]) {
  const float4* wr4 = (const float4*)(W + (size_t)row * N + ncol);
  float4 w0 = wr4[0], w1 = wr4[1], w2 = wr4[2], w3 = wr4[3];
  wa[0] = w0.x; wa[1] = w0.y; wa[2]  = w0.z; wa[3]  = w0.w;
  wa[4] = w1.x; wa[5] = w1.y; wa[6]  = w1.z; wa[7]  = w1.w;
  wa[8] = w2.x; wa[9] = w2.y; wa[10] = w2.z; wa[11] = w2.w;
  wa[12] = w3.x; wa[13] = w3.y; wa[14] = w3.z; wa[15] = w3.w;
}

__device__ __forceinline__ void fetch_x(const float* __restrict__ X, int mrow,
                                        int kk, int khalf, int K, float (&xa)[16]) {
  if (mrow < B_) {
    const float* xr = X + (size_t)mrow * K + kk + khalf * 16;
    const float4* p0 = (const float4*)xr;
    const float4* p1 = (const float4*)(xr + 8);
    float4 a0 = p0[0], a1 = p0[1], b0 = p1[0], b1 = p1[1];
    xa[0] = a0.x; xa[1] = a0.y; xa[2]  = a0.z; xa[3]  = a0.w;
    xa[4] = a1.x; xa[5] = a1.y; xa[6]  = a1.z; xa[7]  = a1.w;
    xa[8] = b0.x; xa[9] = b0.y; xa[10] = b0.z; xa[11] = b0.w;
    xa[12] = b1.x; xa[13] = b1.y; xa[14] = b1.z; xa[15] = b1.w;
  }
}

// ---------------- kernel 1/5: M=8 GEMM  Y(8xN) = X(8xK) @ W(KxN) ----------------
// Explicitly 2x-unrolled software pipeline: two named register sets feed two
// wave-private LDS buffers; the global fetch for tile i+1 is issued before the
// wave fence so HBM loads fly during the WMMA of tile i. No workgroup barriers.
__global__ __launch_bounds__(256)
void gemm8_bf16(const float* __restrict__ X, const float* __restrict__ W,
                float* __restrict__ Y, int K, int N)
{
  __shared__ __bf16 wt[8][2][16][40];   // 20 KB, double buffered per wave
  const int tid   = threadIdx.x;
  const int wave  = tid >> 5;
  const int lane  = tid & 31;
  const int mrow  = lane & 15;
  const int khalf = lane >> 4;
  const int ncol  = blockIdx.x * 128 + wave * 16;

  f32x8 acc = zero8f();

  float wa0[16], xa0[16], wa1[16], xa1[16];
#pragma unroll
  for (int i = 0; i < 16; ++i) { xa0[i] = 0.f; xa1[i] = 0.f; }

  fetch_w(W, lane, ncol, N, wa0);
  fetch_x(X, mrow, 0, khalf, K, xa0);

  for (int kk = 0; kk < K; kk += 64) {    // K is a multiple of 64
    // ---------- stage 0: tile kk (buffer 0) ----------
#pragma unroll
    for (int n = 0; n < 16; ++n) wt[wave][0][n][lane] = (__bf16)wa0[n];
    {
      bf16x16 a = cvt16_pk(xa0);
      __builtin_prefetch(W + (size_t)(kk + 96 + lane) * N + ncol, 0, 1);
      fetch_w(W, kk + 32 + lane, ncol, N, wa1);      // in flight during WMMA
      fetch_x(X, mrow, kk + 32, khalf, K, xa1);
      __builtin_amdgcn_wave_barrier();
      const __bf16* brow = &wt[wave][0][lane & 15][0];
      bf16x8 blo = *(const bf16x8*)(brow + khalf * 16);
      bf16x8 bhi = *(const bf16x8*)(brow + khalf * 16 + 8);
      acc = wmma_bf16(a, pack16(blo, bhi), acc);
    }
    // ---------- stage 1: tile kk+32 (buffer 1) ----------
#pragma unroll
    for (int n = 0; n < 16; ++n) wt[wave][1][n][lane] = (__bf16)wa1[n];
    {
      bf16x16 a = cvt16_pk(xa1);
      if (kk + 64 < K) {
        __builtin_prefetch(W + (size_t)(kk + 128 + lane) * N + ncol, 0, 1);
        fetch_w(W, kk + 64 + lane, ncol, N, wa0);
        fetch_x(X, mrow, kk + 64, khalf, K, xa0);
      }
      __builtin_amdgcn_wave_barrier();
      const __bf16* brow = &wt[wave][1][lane & 15][0];
      bf16x8 blo = *(const bf16x8*)(brow + khalf * 16);
      bf16x8 bhi = *(const bf16x8*)(brow + khalf * 16 + 8);
      acc = wmma_bf16(a, pack16(blo, bhi), acc);
    }
  }

  if (lane < 16) {
#pragma unroll
    for (int r = 0; r < B_; ++r)
      Y[(size_t)r * N + ncol + lane] = acc[r];
  }
}

// ---------------- kernel 2/5: RoPE on Q (B,H*HD) and new K (B,KV*HD) ----------------
__global__ __launch_bounds__(256)
void rope_apply(float* __restrict__ q, float* __restrict__ kn,
                const float* __restrict__ cosv, const float* __restrict__ sinv)
{
  const int QP = B_ * H_ * HD_ / 2;    // 16384 pairs
  const int KP = B_ * KV_ * HD_ / 2;   // 4096 pairs
  int idx = blockIdx.x * 256 + threadIdx.x;
  if (idx < QP) {
    int i = idx & 63;
    float c = cosv[i], s = sinv[i];
    float x0 = q[2 * idx], x1 = q[2 * idx + 1];
    q[2 * idx]     = x0 * c - x1 * s;
    q[2 * idx + 1] = x0 * s + x1 * c;
  } else if (idx < QP + KP) {
    int j = idx - QP;
    int i = j & 63;
    float c = cosv[i], s = sinv[i];
    float x0 = kn[2 * j], x1 = kn[2 * j + 1];
    kn[2 * j]     = x0 * c - x1 * s;
    kn[2 * j + 1] = x0 * s + x1 * c;
  }
}

// ---------------- kernel 3/5: flash-attention chunk ----------------
// grid = (NCHUNK_, B_*KV_) = 64x64. K tile staged fp32 via async global->LDS
// (ASYNCcnt DMA); V staged bf16-transposed concurrently with the K DMA; Q/P
// tiles hold only the 4 real heads (A-fragment row index is mrow&3 — duplicate
// rows land in discarded D rows, so no zero padding needed).
__global__ __launch_bounds__(256)
void attn_flash(const float* __restrict__ q,
                const float* __restrict__ cache_k,
                const float* __restrict__ cache_v,
                const float* __restrict__ k_new,
                const float* __restrict__ v_new,
                const int* __restrict__ sp_ptr,
                float* __restrict__ part_m,
                float* __restrict__ part_l,
                float* __restrict__ part_acc)
{
  __shared__ float  kf[CH_][132];        // fp32 K tile [pos][dim], 33 KB, padded
  __shared__ __bf16 vt[HD_][CH_ + 8];    // V transposed [dim][pos], 18 KB, padded
  __shared__ __bf16 qs[4][136];          // Q heads (bf16), 1.1 KB
  __shared__ __bf16 ps[4][CH_ + 8];      // probabilities, 0.6 KB
  __shared__ float  sc[4][CH_];          // raw scores
  __shared__ float  red[8];              // per-head m, l

  const int chunk = blockIdx.x;
  const int bkv   = blockIdx.y;
  const int b     = bkv >> 3;
  const int kv    = bkv & 7;
  const int sp    = *sp_ptr;             // 4095
  const int pos0  = chunk * CH_;
  const int tid   = threadIdx.x;
  const int wave  = tid >> 5;
  const int lane  = tid & 31;
  const int mrow  = lane & 15;
  const int khalf = lane >> 4;

  // ---- K tile: async DMA fp32 rows (layout-preserving byte copy) ----
  for (int i = tid; i < CH_ * (HD_ / 4); i += 256) {
    int p = i >> 5, d4 = i & 31;
    int pos = pos0 + p;
    const float* src = (pos == sp)
        ? (k_new + (size_t)b * (KV_ * HD_) + kv * HD_)
        : (cache_k + (((size_t)b * MS_ + pos) * KV_ + kv) * HD_);
    const float* g = src + d4 * 4;
    float* l = &kf[p][d4 * 4];
#if ASYNC_LDS
    __builtin_amdgcn_global_load_async_to_lds_b128(
        (gi4_t*)(void*)const_cast<float*>(g), (li4_t*)(void*)l, 0, 0);
#else
    *(float4*)l = *(const float4*)g;
#endif
  }
  // ---- Q heads (bf16) ----
  for (int i = tid; i < 4 * 128; i += 256) {
    int h = i >> 7, d = i & 127;
    qs[h][d] = (__bf16)q[(size_t)b * 4096 + (kv * 4 + h) * 128 + d];
  }
  // ---- V tile transposed (bf16); overlaps the K DMA ----
  for (int i = tid; i < CH_ * HD_; i += 256) {
    int p = i >> 7, d = i & 127;
    int pos = pos0 + p;
    const float* src = (pos == sp)
        ? (v_new + (size_t)b * (KV_ * HD_) + kv * HD_)
        : (cache_v + (((size_t)b * MS_ + pos) * KV_ + kv) * HD_);
    vt[d][p] = (__bf16)src[d];
  }

  WAIT_ASYNC();
  __syncthreads();

  // ---- scores: S = Q(16x128) @ K^T, waves 0..3 own 16 positions each ----
  if (wave < 4) {
    f32x8 s = zero8f();
    const __bf16* arow = &qs[mrow & 3][0];     // rows 4..15 duplicate head rows
    const float*  brow = &kf[wave * 16 + mrow][0];
#pragma unroll
    for (int kk = 0; kk < HD_; kk += 32) {
      bf16x16 a  = pack16(*(const bf16x8*)(arow + kk + khalf * 16),
                          *(const bf16x8*)(arow + kk + khalf * 16 + 8));
      bf16x16 bb = cvt16_f32(brow + kk + khalf * 16);
      s = wmma_bf16(a, bb, s);
    }
    const float scale = 0.088388347648318447f;  // 1/sqrt(128)
    if (lane < 16) {
#pragma unroll
      for (int r = 0; r < 4; ++r) sc[r][wave * 16 + lane] = s[r] * scale;
    }
  }
  __syncthreads();

  // ---- chunk softmax: one wave per head, shuffle reductions ----
  if (wave < 4) {
    int h = wave;
    float v0 = sc[h][lane], v1 = sc[h][lane + 32];
    float mx = fmaxf(v0, v1);
#pragma unroll
    for (int off = 16; off; off >>= 1) mx = fmaxf(mx, __shfl_xor(mx, off));
    float e0 = __expf(v0 - mx), e1 = __expf(v1 - mx);
    ps[h][lane] = (__bf16)e0;
    ps[h][lane + 32] = (__bf16)e1;
    float l = e0 + e1;
#pragma unroll
    for (int off = 16; off; off >>= 1) l += __shfl_xor(l, off);
    if (lane == 0) { red[h] = mx; red[4 + h] = l; }
  }
  __syncthreads();

  // ---- O_partial = P(16x64) @ V(64x128): 8 waves x 16 dims ----
  f32x8 o = zero8f();
  {
    const __bf16* arow = &ps[mrow & 3][0];     // duplicate rows -> discarded D rows
    const __bf16* brow = &vt[wave * 16 + mrow][0];
#pragma unroll
    for (int kk = 0; kk < CH_; kk += 32) {
      bf16x16 a  = pack16(*(const bf16x8*)(arow + kk + khalf * 16),
                          *(const bf16x8*)(arow + kk + khalf * 16 + 8));
      bf16x16 bb = pack16(*(const bf16x8*)(brow + kk + khalf * 16),
                          *(const bf16x8*)(brow + kk + khalf * 16 + 8));
      o = wmma_bf16(a, bb, o);
    }
  }
  const size_t base = ((size_t)bkv * NCHUNK_ + chunk) * 4;
  if (lane < 16) {
#pragma unroll
    for (int r = 0; r < 4; ++r)
      part_acc[(base + r) * 128 + wave * 16 + lane] = o[r];
  }
  if (tid < 4) { part_m[base + tid] = red[tid]; part_l[base + tid] = red[4 + tid]; }
}

// ---------------- kernel 4/5: merge chunk partials (log-sum-exp) ----------------
__global__ __launch_bounds__(256)
void attn_combine(const float* __restrict__ pm, const float* __restrict__ pl,
                  const float* __restrict__ pa, float* __restrict__ ao)
{
  __shared__ float M[4], L[4];
  const int bkv = blockIdx.x, b = bkv >> 3, kv = bkv & 7;
  const int tid = threadIdx.x;
  if (tid < 4) {
    float mm = -1e30f;
    for (int c = 0; c < NCHUNK_; ++c)
      mm = fmaxf(mm, pm[((size_t)bkv * NCHUNK_ + c) * 4 + tid]);
    float l = 0.f;
    for (int c = 0; c < NCHUNK_; ++c) {
      size_t idx = ((size_t)bkv * NCHUNK_ + c) * 4 + tid;
      l += pl[idx] * __expf(pm[idx] - mm);
    }
    M[tid] = mm; L[tid] = l;
  }
  __syncthreads();
  for (int i = tid; i < 4 * 128; i += 256) {
    int h = i >> 7, d = i & 127;
    float acc = 0.f;
    for (int c = 0; c < NCHUNK_; ++c) {
      size_t idx = ((size_t)bkv * NCHUNK_ + c) * 4 + h;
      acc += pa[idx * 128 + d] * __expf(pm[idx] - M[h]);
    }
    ao[(size_t)b * 4096 + (kv * 4 + h) * 128 + d] = acc / L[h];
  }
}

// ---------------- launch ----------------
extern "C" void kernel_launch(void* const* d_in, const int* in_sizes, int n_in,
                              void* d_out, int out_size, void* d_ws, size_t ws_size,
                              hipStream_t stream) {
  const float* x   = (const float*)d_in[0];
  const float* wq  = (const float*)d_in[1];
  const float* wk  = (const float*)d_in[2];
  const float* wv  = (const float*)d_in[3];
  const float* wo  = (const float*)d_in[4];
  const float* ck  = (const float*)d_in[5];
  const float* cv  = (const float*)d_in[6];
  const float* fc  = (const float*)d_in[7];
  const float* fs  = (const float*)d_in[8];
  const int*   sp  = (const int*)d_in[9];
  float* out = (float*)d_out;

  float* ws = (float*)d_ws;
  float* q_ws  = ws;                  // 8*4096
  float* kn_ws = q_ws  + 32768;       // 8*1024
  float* vn_ws = kn_ws + 8192;        // 8*1024
  float* ao_ws = vn_ws + 8192;        // 8*4096
  float* pm_ws = ao_ws + 32768;       // 64*64*4
  float* pl_ws = pm_ws + 16384;       // 64*64*4
  float* pa_ws = pl_ws + 16384;       // 64*64*4*128

  gemm8_bf16<<<32, 256, 0, stream>>>(x, wq, q_ws, D_, H_ * HD_);
  gemm8_bf16<<<8,  256, 0, stream>>>(x, wk, kn_ws, D_, KV_ * HD_);
  gemm8_bf16<<<8,  256, 0, stream>>>(x, wv, vn_ws, D_, KV_ * HD_);

  rope_apply<<<(16384 + 4096 + 255) / 256, 256, 0, stream>>>(q_ws, kn_ws, fc, fs);

  attn_flash<<<dim3(NCHUNK_, B_ * KV_), 256, 0, stream>>>(
      q_ws, ck, cv, kn_ws, vn_ws, sp, pm_ws, pl_ws, pa_ws);

  attn_combine<<<B_ * KV_, 256, 0, stream>>>(pm_ws, pl_ws, pa_ws, ao_ws);

  gemm8_bf16<<<32, 256, 0, stream>>>(ao_ws, wo, out, D_, D_);
}
